// GSA_actor_85272280695433
// MI455X (gfx1250) — matrix operand order repaired and checked by reading, not compiled, not run on hardware
//
#include <hip/hip_runtime.h>
#include <math.h>

typedef _Float16 h16;
typedef __attribute__((ext_vector_type(16))) _Float16 v16h;
typedef __attribute__((ext_vector_type(8)))  float    v8f;

__device__ __forceinline__ float fast_rcp(float v){ return __builtin_amdgcn_rcpf(v); }
__device__ __forceinline__ float silu_f(float v){ return v * fast_rcp(1.0f + __expf(-v)); }

__device__ __forceinline__ v8f wmma16(v16h a, v16h b, v8f c){
    // D = A(16x32 f16) x B(32x16 f16) + C(16x16 f32)
    return __builtin_amdgcn_wmma_f32_16x16x32_f16(false, a, false, b, (short)0, c, false, false);
}

// ---- operand packing -------------------------------------------------------
// CDNA5 16-bit A layout: lanes 0-15 hold K {0..7,16..23}, lanes 16-31 hold K {8..15,24..31}; M = lane%16.
// CDNA5 B layout: same K order as A (lanes 0-15 hold rows k {0..7,16..23}, lanes 16-31 k {8..15,24..31}); N = lane%16.

// A from row-major [32][16] source (K<=16, upper K zero-filled): one contiguous 16B run per lane.
__device__ __forceinline__ v16h pack_A16(const h16* s, int lane, int mbase){
    v16h a;
    int m = mbase + (lane & 15);
    int khi = lane >> 4;
    #pragma unroll
    for (int j = 0; j < 16; ++j){
        int k = (j & 7) + 8*khi + 16*(j >> 3);
        a[j] = (k < 16) ? s[m*16 + k] : (h16)0.0f;
    }
    return a;
}

// A from transposed [16][32] source: A[m,k] = s[m*32+k], full K=32: two contiguous 16B runs per lane.
__device__ __forceinline__ v16h pack_AT32(const h16* s, int lane){
    v16h a;
    int m = lane & 15;
    int khi = lane >> 4;
    #pragma unroll
    for (int j = 0; j < 16; ++j){
        int k = (j & 7) + 8*khi + 16*(j >> 3);
        a[j] = s[m*32 + k];
    }
    return a;
}

// B from transposed [16][32] source: B[k,n] = s[n*32+k]: two contiguous 16B runs per lane (table k order).
__device__ __forceinline__ v16h pack_B32(const h16* s, int lane){
    v16h b;
    int n = lane & 15;
    int khi = lane >> 4;
    #pragma unroll
    for (int j = 0; j < 16; ++j){
        b[j] = s[n*32 + (j & 7) + 8*khi + 16*(j >> 3)];
    }
    return b;
}

// C (f32) -> row-major [32][16] f16 with LDS bias (zero-padded) + silu, branchless.
// C layout: N = lane%16, row m = r + 8*(lane/16).
__device__ __forceinline__ void store_C_silu(const v8f c, h16* d, int lane, int mbase,
                                             const float* biasLds){
    int n = lane & 15;
    float bv = biasLds[n];
    #pragma unroll
    for (int r = 0; r < 8; ++r){
        int m = mbase + r + 8*(lane >> 4);
        d[m*16 + n] = (h16)silu_f(c[r] + bv);
    }
}

// C (f32) -> transposed [16][32] f16 (row = output dim n, col = agent m).
__device__ __forceinline__ void store_C_siluT(const v8f c, h16* d, int lane, int mbase,
                                              const float* biasLds){
    int n = lane & 15;
    float bv = biasLds[n];
    #pragma unroll
    for (int r = 0; r < 8; ++r){
        int m = mbase + r + 8*(lane >> 4);
        d[n*32 + m] = (h16)silu_f(c[r] + bv);
    }
}

template<int MODE>
__global__ __launch_bounds__(128)
void gsa_attn(const float* __restrict__ x,
              const float* __restrict__ Win, const float* __restrict__ bin,
              const float* __restrict__ Aq1, const float* __restrict__ Ak1, const float* __restrict__ Av1,
              const float* __restrict__ Bq1, const float* __restrict__ Bk1, const float* __restrict__ Bv1,
              const float* __restrict__ Wh,  const float* __restrict__ bh,
              const float* __restrict__ Aq2, const float* __restrict__ Ak2, const float* __restrict__ Av2,
              const float* __restrict__ Bq2, const float* __restrict__ Bk2, const float* __restrict__ Bv2,
              const float* __restrict__ Wout,const float* __restrict__ bout,
              const float* __restrict__ umean, float* __restrict__ out)
{
    constexpr int IN_DIM = MODE ? 8 : 6;

    __shared__ float sx[32*6];        // x[b]
    __shared__ float sL[32*32];       // gate matrix
    __shared__ float sP[32*16];       // x @ Win_obs^T, zero-padded stride 16
    __shared__ float sQm[32*16];      // umean @ Win_act^T, zero-padded (std net only)
    __shared__ float sBias[8][16];    // Bq1,Bk1,Bv1,bh,Bq2,Bk2,Bv2,zeros — zero padded
    __shared__ float sbin[16];        // bin zero-padded
    __shared__ float sHead[16];       // Wout(12) + bout(2) + pad
    __shared__ h16   wBT[7][512];     // q1,k1,v1,wh,q2,k2,v2 as TRANSPOSED B operands [16][32]
    __shared__ h16   bufs[4][6][512]; // per-wave: X1/H[32][16], QT[16][32], KT[16][32],
                                      //           V[32][16],  ST[16][32], Z[32][16]
    __shared__ float smean[4][16];

    const int tid  = threadIdx.x;
    const int lane = tid & 31;
    const int wid  = tid >> 5;
    const int b    = blockIdx.x;

    for (int idx = tid; idx < 32*6; idx += 128) sx[idx] = x[b*32*6 + idx];
    __syncthreads();

    for (int idx = tid; idx < 32*32; idx += 128){
        int i = idx >> 5, j = idx & 31;
        float dx = sx[i*6+0] - sx[j*6+0];
        float dy = sx[i*6+1] - sx[j*6+1];
        float dist = sqrtf(dx*dx + dy*dy);
        sL[idx] = (dist <= 1.0f) ? fast_rcp(1.0f + __expf(2.0f*(dist - 1.0f))) : 0.0f;
    }
    for (int idx = tid; idx < 32*16; idx += 128){
        int j = idx >> 4, d = idx & 15;
        float acc = 0.0f;
        if (d < 12){
            #pragma unroll
            for (int k = 0; k < 6; ++k) acc += Win[d*IN_DIM + k]*sx[j*6+k];
        }
        sP[idx] = acc;
        float qm = 0.0f;
        if (MODE && d < 12)
            qm = Win[d*8+6]*umean[(b*32+j)*2+0] + Win[d*8+7]*umean[(b*32+j)*2+1];
        sQm[idx] = qm;
    }
    if (tid < 128){
        int grp = tid >> 4, n = tid & 15;
        float v = 0.0f;
        if      (grp == 0){ if (n < 12) v = Bq1[n]; }
        else if (grp == 1){ if (n < 12) v = Bk1[n]; }
        else if (grp == 2){ if (n < 12) v = Bv1[n]; }
        else if (grp == 3){ if (n < 6 ) v = bh [n]; }
        else if (grp == 4){ if (n < 6 ) v = Bq2[n]; }
        else if (grp == 5){ if (n < 6 ) v = Bk2[n]; }
        else if (grp == 6){ if (n < 6 ) v = Bv2[n]; }
        sBias[grp][n] = v;
        if (grp == 0) sbin[n]  = (n < 12) ? bin[n] : 0.0f;
        if (grp == 1) sHead[n] = (n < 12) ? Wout[n] : ((n < 14) ? bout[n-12] : 0.0f);
    }
    // transposed weight-B operands: wBT[w][d*32 + e] = B[k=e][n=d]
    for (int idx = tid; idx < 512; idx += 128){
        int d = idx >> 5, e = idx & 31;
        wBT[0][idx] = (d<12 && e<12) ? (h16)Aq1[d*12+e] : (h16)0.0f;
        wBT[1][idx] = (d<12 && e<12) ? (h16)Ak1[d*12+e] : (h16)0.0f;
        wBT[2][idx] = (d<12 && e<12) ? (h16)Av1[d*12+e] : (h16)0.0f;
        wBT[3][idx] = (d<6  && e<12) ? (h16)Wh [d*12+e] : (h16)0.0f;   // B[k=dz][n=h] = Wh[h][dz]
        wBT[4][idx] = (d<6  && e<6 ) ? (h16)Aq2[d*6 +e] : (h16)0.0f;
        wBT[5][idx] = (d<6  && e<6 ) ? (h16)Ak2[d*6 +e] : (h16)0.0f;
        wBT[6][idx] = (d<6  && e<6 ) ? (h16)Av2[d*6 +e] : (h16)0.0f;
    }
    // ST pad columns e=16..31 must stay zero (K padding for the SV GEMM)
    for (int idx = tid; idx < 4*256; idx += 128){
        int w = idx >> 8, o = idx & 255;
        bufs[w][4][(o >> 4)*32 + 16 + (o & 15)] = (h16)0.0f;
    }
    __syncthreads();

    h16* X1  = bufs[wid][0];   // [32][16] A-operand
    h16* QbT = bufs[wid][1];   // [16][32]
    h16* KbT = bufs[wid][2];   // [16][32]
    h16* Vb  = bufs[wid][3];   // [32][16] A-operand
    h16* SbT = bufs[wid][4];   // [16][32]
    h16* Zb  = bufs[wid][5];   // [32][16] A-operand

    for (int t = 0; t < 8; ++t){
        const int i = wid*8 + t;

        // X1[j][d] = silu(L[i][j]*P[j][d] (+ Qm[i][d]) + bin[d]); branchless, zero padded
        #pragma unroll
        for (int rep = 0; rep < 16; ++rep){
            int e = rep*32 + lane;
            int j = e >> 4, d = e & 15;
            float v = sL[i*32 + j]*sP[j*16 + d] + sbin[d];
            if (MODE) v += sQm[i*16 + d];
            X1[e] = (h16)silu_f(v);
        }
        __syncthreads();

        const h16* cur = X1;
        #pragma unroll
        for (int blk = 0; blk < 2; ++blk){
            const h16*   WQb = blk ? wBT[4] : wBT[0];
            const h16*   WKb = blk ? wBT[5] : wBT[1];
            const h16*   WVb = blk ? wBT[6] : wBT[2];
            const float* BQb = blk ? sBias[4] : sBias[0];
            const float* BKb = blk ? sBias[5] : sBias[1];
            const float* BVb = blk ? sBias[6] : sBias[2];
            const int    dim = blk ? 6 : 12;

            // ---- Q/K/V projections: out[n][d] = silu(sum_e cur[n][e]*A[d][e] + B[d]) ----
            {
                v16h bq = pack_B32(WQb, lane);
                v16h bk = pack_B32(WKb, lane);
                v16h bv = pack_B32(WVb, lane);
                #pragma unroll
                for (int mb = 0; mb < 32; mb += 16){
                    v16h a = pack_A16(cur, lane, mb);
                    v8f c0 = {0.f,0.f,0.f,0.f,0.f,0.f,0.f,0.f};
                    v8f cq = wmma16(a, bq, c0);
                    store_C_siluT(cq, QbT, lane, mb, BQb);   // QT[d][agent]
                    v8f ck = wmma16(a, bk, c0);
                    store_C_siluT(ck, KbT, lane, mb, BKb);   // KT[d][agent]
                    v8f cv = wmma16(a, bv, c0);
                    store_C_silu (cv, Vb,  lane, mb, BVb);   // V[agent][d]
                }
            }
            __syncthreads();

            // ---- S = softmax_rows(Q K^T): S[d,e] = sum_n QT[d][n]*KT[e][n], K=32 exact ----
            {
                v16h a  = pack_AT32(QbT, lane);  // A[m=d, k=agent]
                v16h bb = pack_B32 (KbT, lane);  // B[k=agent, n=e]
                v8f  c0 = {0.f,0.f,0.f,0.f,0.f,0.f,0.f,0.f};
                v8f  c  = wmma16(a, bb, c0);
                int n = lane & 15, half = lane >> 4;
                #pragma unroll
                for (int r = 0; r < 8; ++r){
                    float v = (n < dim) ? c[r] : -1e30f;
                    float mx = v;
                    mx = fmaxf(mx, __shfl_xor(mx, 1));
                    mx = fmaxf(mx, __shfl_xor(mx, 2));
                    mx = fmaxf(mx, __shfl_xor(mx, 4));
                    mx = fmaxf(mx, __shfl_xor(mx, 8));
                    float ev = __expf(v - mx);
                    float sm = ev;
                    sm += __shfl_xor(sm, 1);
                    sm += __shfl_xor(sm, 2);
                    sm += __shfl_xor(sm, 4);
                    sm += __shfl_xor(sm, 8);
                    float s = ev * fast_rcp(sm);
                    int d = r + 8*half;
                    // ST[d][e] = S[d][e]  (transposed B operand for the SV GEMM)
                    SbT[d*32 + n] = (n < dim && d < dim) ? (h16)s : (h16)0.0f;
                }
            }
            __syncthreads();

            if (blk == 0){
                // ---- Z[n][d] = silu(sum_e S[d,e]*V[n][e]) ----
                v16h bS = pack_B32(SbT, lane);   // B[k=e, n=d]
                #pragma unroll
                for (int mb = 0; mb < 32; mb += 16){
                    v16h a = pack_A16(Vb, lane, mb);
                    v8f c0 = {0.f,0.f,0.f,0.f,0.f,0.f,0.f,0.f};
                    v8f c  = wmma16(a, bS, c0);
                    store_C_silu(c, Zb, lane, mb, sBias[7]);   // zero bias row
                }
                __syncthreads();
                // ---- H[n][h] = silu(sum_d Z[n][d]*Wh[h][d] + bh[h]) -> reuse X1 ----
                v16h bw = pack_B32(wBT[3], lane);
                #pragma unroll
                for (int mb = 0; mb < 32; mb += 16){
                    v16h a = pack_A16(Zb, lane, mb);
                    v8f c0 = {0.f,0.f,0.f,0.f,0.f,0.f,0.f,0.f};
                    v8f c  = wmma16(a, bw, c0);
                    store_C_silu(c, X1, lane, mb, sBias[3]);
                }
                __syncthreads();
                cur = X1;
            } else {
                // ---- final: mean_n silu(sum_e S2[d,e]*V2[n][e]) -> Wout head ----
                v16h bS = pack_B32(SbT, lane);
                float colsum = 0.0f;
                #pragma unroll
                for (int mb = 0; mb < 32; mb += 16){
                    v16h a = pack_A16(Vb, lane, mb);
                    v8f c0 = {0.f,0.f,0.f,0.f,0.f,0.f,0.f,0.f};
                    v8f c  = wmma16(a, bS, c0);
                    #pragma unroll
                    for (int r = 0; r < 8; ++r) colsum += silu_f(c[r]);
                }
                colsum += __shfl_xor(colsum, 16);
                if (lane < 16) smean[wid][lane] = colsum * (1.0f/32.0f);
                __syncthreads();
                if (lane < 2){
                    float acc = sHead[12 + lane];
                    #pragma unroll
                    for (int d2 = 0; d2 < 6; ++d2) acc += sHead[lane*6 + d2]*smean[wid][d2];
                    float o = silu_f(acc);
                    if (MODE){
                        float th = tanhf(o);
                        o = __expf(-5.0f + 3.5f*(th + 1.0f));  // exp(LOG_STD_MIN + 0.5*(MAX-MIN)*(tanh+1))
                    }
                    out[(b*32 + i)*2 + lane] = o;
                }
                __syncthreads();
            }
        }
    }
}

extern "C" void kernel_launch(void* const* d_in, const int* in_sizes, int n_in,
                              void* d_out, int out_size, void* d_ws, size_t ws_size,
                              hipStream_t stream) {
    (void)in_sizes; (void)n_in; (void)d_ws; (void)ws_size; (void)out_size;
    const float* x = (const float*)d_in[0];
    float* out = (float*)d_out;
    #define P(i) ((const float*)d_in[i])
    // pass 1: u_mean -> out[0 .. 131071]
    gsa_attn<0><<<dim3(2048), dim3(128), 0, stream>>>(x,
        P(1),P(2),P(3),P(4),P(5),P(6),P(7),P(8),P(9),P(10),
        P(11),P(12),P(13),P(14),P(15),P(16),P(17),P(18),
        nullptr, out);
    // pass 2: u_std -> out[131072 .. 262143], consumes u_mean
    gsa_attn<1><<<dim3(2048), dim3(128), 0, stream>>>(x,
        P(19),P(20),P(21),P(22),P(23),P(24),P(25),P(26),P(27),P(28),
        P(29),P(30),P(31),P(32),P(33),P(34),P(35),P(36),
        out, out + 2048*32*2);
    #undef P
}
